// LSTM_11149735100706
// MI455X (gfx1250) — matrix operand order repaired and checked by reading, not compiled, not run on hardware
//
#include <hip/hip_runtime.h>

// LSTM scan on gfx1250: f16 WMMA gate GEMMs fused with cell update.
// Sizes from reference: B=256, SEQ=512, D_IN=256, H=1024, NCLS=128.

typedef __attribute__((ext_vector_type(16))) _Float16 v16h;
typedef __attribute__((ext_vector_type(8)))  float    v8f;

#define SEQ   512
#define DIN   256
#define HID   1024
#define NCLS  128
#define BATCH 256
#define KTOT  1280          // DIN + HID (fused K dimension)
#define NG    4096          // 4 gates * HID

// byte stride between gate blocks in the fused, column-major weight matrix
#define GATE_STRIDE ((size_t)HID * KTOT * 2)   // 2,621,440 B; gate3 fits 24-bit imm

union Frag16 { v16h v; int4 q[2]; };

__device__ __forceinline__ float sigmoidf_(float x) {
    return 1.0f / (1.0f + __expf(-x));
}

// ---------------- setup kernels ----------------

__global__ void k_f32_to_f16(const float* __restrict__ src,
                             _Float16* __restrict__ dst, int n) {
    int i = blockIdx.x * blockDim.x + threadIdx.x;
    if (i < n) dst[i] = (_Float16)src[i];
}

// Fused weight matrix, column-major (K contiguous per column).
// grid = (5, 4096): blockIdx.y = column n (gate*1024+j), k = blockIdx.x*256+tid.
__global__ void k_build_wcat(const float* __restrict__ Wfx, const float* __restrict__ Wfh,
                             const float* __restrict__ Wix, const float* __restrict__ Wih,
                             const float* __restrict__ Wgx, const float* __restrict__ Wgh,
                             const float* __restrict__ Wox, const float* __restrict__ Woh,
                             _Float16* __restrict__ wcat) {
    const int n = blockIdx.y;                       // 0..4095
    const int k = blockIdx.x * 256 + threadIdx.x;   // 0..1279
    if (k >= KTOT) return;
    const int gate = n >> 10;
    const int j = n & 1023;
    const float* Wx; const float* Wh;
    switch (gate) {
        case 0:  Wx = Wfx; Wh = Wfh; break;
        case 1:  Wx = Wix; Wh = Wih; break;
        case 2:  Wx = Wgx; Wh = Wgh; break;
        default: Wx = Wox; Wh = Woh; break;
    }
    const float v = (k < DIN) ? Wx[(size_t)k * HID + j]
                              : Wh[(size_t)(k - DIN) * HID + j];
    wcat[(size_t)n * KTOT + k] = (_Float16)v;
}

// Wph (H x NCLS row-major) -> column-major f16. grid = (4, 128).
__global__ void k_build_wph(const float* __restrict__ Wph, _Float16* __restrict__ wph16) {
    const int n = blockIdx.y;                       // 0..127
    const int k = blockIdx.x * 256 + threadIdx.x;   // 0..1023
    wph16[(size_t)n * HID + k] = (_Float16)Wph[(size_t)k * NCLS + n];
}

__global__ void k_init_state(float* __restrict__ C, _Float16* __restrict__ h0) {
    int i = blockIdx.x * blockDim.x + threadIdx.x;
    if (i < BATCH * HID) { C[i] = 0.0f; h0[i] = (_Float16)0.0f; }
}

// ---------------- per-timestep kernel ----------------
// Grid: (16, 4) workgroups of 256 threads; workgroup tile = 64 batch rows x 64 hidden cols.
// 8 waves: 2 (M) x 4 (J); each wave owns a 32x16 h-tile: two M-subtiles x 4 gates
// (8 accumulators), so every B fragment feeds two WMMAs.
// A = [emb[tok] | h_prev] staged via LDS in 256-wide K blocks.

__global__ __launch_bounds__(256) void k_lstm_step(
    const int* __restrict__ x, const _Float16* __restrict__ emb16,
    const _Float16* __restrict__ wcat, const _Float16* __restrict__ hsrc,
    const float* __restrict__ bf, const float* __restrict__ bi,
    const float* __restrict__ bg, const float* __restrict__ bo,
    float* __restrict__ C, _Float16* __restrict__ hdst, int t)
{
    __shared__ _Float16 As[64][272];          // 64 rows x 256 K (+16 pad, keeps 16B align)

    const int tid   = threadIdx.x;
    const int lane  = tid & 31;
    const int wave  = tid >> 5;
    const int waveM = wave >> 2;              // 0..1  (32 rows each)
    const int waveJ = wave & 3;               // 0..3
    const int wgJ   = blockIdx.x;             // 0..15
    const int wgM   = blockIdx.y;             // 0..3  (64-row blocks)

    const int ln16 = lane & 15;
    const int hsel = lane >> 4;               // 0 or 1 (lane half)

    const int jn  = wgJ * 64 + waveJ * 16 + ln16;  // hidden column for B/C/D lanes
    const int mr0 = waveM * 32 + ln16;             // LDS row of A fragment 0 (frag 1 = +16)

    // Single 64-bit B base per wave; all gate / K offsets become load immediates.
    // B frag layout: lanes 0-15 hold K=0..15, lanes 16-31 hold K=16..31 (hsel*32 bytes).
    const char* bcol = (const char*)(wcat + (size_t)jn * KTOT) + hsel * 32;

    // cooperative-fill coordinates: 4 threads per row, 32 halves (4 x int4) each
    const int fr = tid >> 2;                  // 0..63
    const int fk = (tid & 3) * 64;            // 0,64,128,192

    v8f aF0 = {}, aI0 = {}, aG0 = {}, aO0 = {};
    v8f aF1 = {}, aI1 = {}, aG1 = {}, aO1 = {};

    for (int kb = 0; kb < 5; ++kb) {
        // ---- stage A block (64 x 256 f16) into LDS ----
        {
            const int grow = wgM * 64 + fr;
            const _Float16* src;
            if (kb == 0) {
                const int tok = x[grow * SEQ + t];
                src = emb16 + (size_t)tok * DIN + fk;               // embedding gather
            } else {
                src = hsrc + (size_t)grow * HID + (kb * 256 - DIN) + fk;  // previous h
            }
            const int4* s4 = (const int4*)src;
            int4* d4 = (int4*)&As[fr][fk];
            d4[0] = s4[0]; d4[1] = s4[1]; d4[2] = s4[2]; d4[3] = s4[3];
        }
        __syncthreads();

        const char* bk = bcol + kb * 512;     // advance along K (256 halves)

        // ---- 8 K-chunks of 32; 2 A frags x 4 gate B frags -> 8 WMMAs ----
#pragma unroll
        for (int kc = 0; kc < 8; ++kc) {
            const int k0 = kc * 32;
            Frag16 a0, a1;
            // A frag K order used here, same as B: lane half selects contiguous K range [hsel*16,+16)
            a0.q[0] = *(const int4*)&As[mr0][k0 + hsel * 16];
            a0.q[1] = *(const int4*)&As[mr0][k0 + hsel * 16 + 8];
            a1.q[0] = *(const int4*)&As[mr0 + 16][k0 + hsel * 16];
            a1.q[1] = *(const int4*)&As[mr0 + 16][k0 + hsel * 16 + 8];

            const int ko = k0 * 2;            // byte offset within K block
            Frag16 b0, b1, b2, b3;
            b0.q[0] = *(const int4*)(bk + 0 * GATE_STRIDE + ko);
            b0.q[1] = *(const int4*)(bk + 0 * GATE_STRIDE + ko + 16);
            b1.q[0] = *(const int4*)(bk + 1 * GATE_STRIDE + ko);
            b1.q[1] = *(const int4*)(bk + 1 * GATE_STRIDE + ko + 16);
            b2.q[0] = *(const int4*)(bk + 2 * GATE_STRIDE + ko);
            b2.q[1] = *(const int4*)(bk + 2 * GATE_STRIDE + ko + 16);
            b3.q[0] = *(const int4*)(bk + 3 * GATE_STRIDE + ko);
            b3.q[1] = *(const int4*)(bk + 3 * GATE_STRIDE + ko + 16);

            aF0 = __builtin_amdgcn_wmma_f32_16x16x32_f16(false, a0.v, false, b0.v, (short)0, aF0, false, false);
            aF1 = __builtin_amdgcn_wmma_f32_16x16x32_f16(false, a1.v, false, b0.v, (short)0, aF1, false, false);
            aI0 = __builtin_amdgcn_wmma_f32_16x16x32_f16(false, a0.v, false, b1.v, (short)0, aI0, false, false);
            aI1 = __builtin_amdgcn_wmma_f32_16x16x32_f16(false, a1.v, false, b1.v, (short)0, aI1, false, false);
            aG0 = __builtin_amdgcn_wmma_f32_16x16x32_f16(false, a0.v, false, b2.v, (short)0, aG0, false, false);
            aG1 = __builtin_amdgcn_wmma_f32_16x16x32_f16(false, a1.v, false, b2.v, (short)0, aG1, false, false);
            aO0 = __builtin_amdgcn_wmma_f32_16x16x32_f16(false, a0.v, false, b3.v, (short)0, aO0, false, false);
            aO1 = __builtin_amdgcn_wmma_f32_16x16x32_f16(false, a1.v, false, b3.v, (short)0, aO1, false, false);
        }
        __syncthreads();
    }

    // ---- fused epilogue: gates -> cell update -> h (fp32 math) ----
    const float bfv = bf[jn], biv = bi[jn], bgv = bg[jn], bov = bo[jn];
#pragma unroll
    for (int sub = 0; sub < 2; ++sub) {
        const v8f& vF = sub ? aF1 : aF0;
        const v8f& vI = sub ? aI1 : aI0;
        const v8f& vG = sub ? aG1 : aG0;
        const v8f& vO = sub ? aO1 : aO0;
#pragma unroll
        for (int r = 0; r < 8; ++r) {
            // C/D layout: VGPR r holds M = r + 8*hsel, N = ln16
            const int m = wgM * 64 + waveM * 32 + sub * 16 + hsel * 8 + r;
            const float fg = sigmoidf_(vF[r] + bfv);
            const float ig = sigmoidf_(vI[r] + biv);
            const float gg = sigmoidf_(vG[r] + bgv);   // reference uses sigmoid for g
            const float og = sigmoidf_(vO[r] + bov);
            const size_t off = (size_t)m * HID + jn;
            float Cn = gg * ig + C[off] * fg;
            const int tok = x[m * SEQ + t];
            if (tok <= 0) Cn = 0.0f;
            C[off] = Cn;
            hdst[off] = (_Float16)(og * tanhf(Cn));
        }
    }
}

// ---------------- final projection: y = h @ Wph + bp ----------------
// Grid (2, 8), 8 waves per block, wave tile 16x16, K=1024 (32 WMMAs).

__global__ __launch_bounds__(256) void k_proj(
    const _Float16* __restrict__ h16, const _Float16* __restrict__ wph16,
    const float* __restrict__ bp, float* __restrict__ y)
{
    const int tid   = threadIdx.x;
    const int lane  = tid & 31;
    const int wave  = tid >> 5;
    const int waveM = wave >> 2;
    const int waveN = wave & 3;
    const int wgN   = blockIdx.x;  // 0..1
    const int wgM   = blockIdx.y;  // 0..7
    const int ln16  = lane & 15;
    const int hsel  = lane >> 4;

    const int n  = wgN * 64 + waveN * 16 + ln16;
    const int mA = wgM * 32 + waveM * 16 + ln16;

    const char* abase = (const char*)(h16 + (size_t)mA * HID) + hsel * 32;
    const char* bbase = (const char*)(wph16 + (size_t)n * HID) + hsel * 32;

    v8f acc = {};
#pragma unroll 8
    for (int kc = 0; kc < 32; ++kc) {
        const int ko = kc * 64;     // 32 halves per chunk
        Frag16 a, b;
        a.q[0] = *(const int4*)(abase + ko);
        a.q[1] = *(const int4*)(abase + ko + 16);
        b.q[0] = *(const int4*)(bbase + ko);
        b.q[1] = *(const int4*)(bbase + ko + 16);
        acc = __builtin_amdgcn_wmma_f32_16x16x32_f16(false, a.v, false, b.v,
                                                     (short)0, acc, false, false);
    }
    const float bias = bp[n];
#pragma unroll
    for (int r = 0; r < 8; ++r) {
        const int m = wgM * 32 + waveM * 16 + hsel * 8 + r;
        y[(size_t)m * NCLS + n] = acc[r] + bias;
    }
}

// ---------------- launcher ----------------

extern "C" void kernel_launch(void* const* d_in, const int* in_sizes, int n_in,
                              void* d_out, int out_size, void* d_ws, size_t ws_size,
                              hipStream_t stream) {
    (void)in_sizes; (void)n_in; (void)out_size; (void)ws_size;

    const int*   x   = (const int*)  d_in[0];
    const float* emb = (const float*)d_in[1];
    const float* Wfx = (const float*)d_in[2];
    const float* Wfh = (const float*)d_in[3];
    const float* bf  = (const float*)d_in[4];
    const float* Wix = (const float*)d_in[5];
    const float* Wih = (const float*)d_in[6];
    const float* bi  = (const float*)d_in[7];
    const float* Wgx = (const float*)d_in[8];
    const float* Wgh = (const float*)d_in[9];
    const float* bg  = (const float*)d_in[10];
    const float* Wox = (const float*)d_in[11];
    const float* Woh = (const float*)d_in[12];
    const float* bo  = (const float*)d_in[13];
    const float* Wph = (const float*)d_in[14];
    const float* bp  = (const float*)d_in[15];
    float* y = (float*)d_out;

    // workspace layout (all offsets 512B-aligned), total ~12.9 MB
    char* ws = (char*)d_ws;
    _Float16* emb16 = (_Float16*)(ws + 0);          //   66,560 B (130*256 f16)
    _Float16* wcat  = (_Float16*)(ws + 66560);      // 10,485,760 B (1280*4096 f16, col-major)
    _Float16* wph16 = (_Float16*)(ws + 10552320);   //    262,144 B (128 cols * 1024 K f16)
    float*    Cbuf  = (float*)   (ws + 10814464);   //  1,048,576 B (256*1024 f32)
    _Float16* hbuf0 = (_Float16*)(ws + 11863040);   //    524,288 B
    _Float16* hbuf1 = (_Float16*)(ws + 12387328);   //    524,288 B

    const int embN = (NCLS + 2) * DIN;  // 130*256
    k_f32_to_f16<<<(embN + 255) / 256, 256, 0, stream>>>(emb, emb16, embN);
    k_build_wcat<<<dim3(5, NG), 256, 0, stream>>>(
        Wfx, Wfh, Wix, Wih, Wgx, Wgh, Wox, Woh, wcat);
    k_build_wph<<<dim3(4, NCLS), 256, 0, stream>>>(Wph, wph16);
    k_init_state<<<(BATCH * HID + 255) / 256, 256, 0, stream>>>(Cbuf, hbuf0);

    dim3 gridStep(16, 4);   // 64 WGs: J tiles of 64 x M tiles of 64
    for (int t = 0; t < SEQ; ++t) {
        const _Float16* hs = (t & 1) ? hbuf1 : hbuf0;
        _Float16*       hd = (t & 1) ? hbuf0 : hbuf1;
        k_lstm_step<<<gridStep, 256, 0, stream>>>(x, emb16, wcat, hs,
                                                  bf, bi, bg, bo, Cbuf, hd, t);
    }
    // SEQ=512 is even: final h landed in hbuf0
    dim3 gridProj(2, 8);
    k_proj<<<gridProj, 256, 0, stream>>>(hbuf0, wph16, bp, y);
}
